// PPO_29575144800760
// MI455X (gfx1250) — matrix operand order repaired and checked by reading, not compiled, not run on hardware
//
#include <hip/hip_runtime.h>
#include <hip/hip_bf16.h>

typedef _Float16 fp16;
typedef __attribute__((ext_vector_type(8)))  _Float16 h8v;
typedef __attribute__((ext_vector_type(16))) _Float16 h16v;
typedef __attribute__((ext_vector_type(8)))  float    f8v;

#define HD 256
#define NHEAD 4
#define NEG_SLOPE 0.2f

// ---- orderable-uint encoding for float atomic max -------------------------
__device__ __forceinline__ unsigned fenc(float f) {
  unsigned u = __float_as_uint(f);
  return (u & 0x80000000u) ? ~u : (u | 0x80000000u);
}
__device__ __forceinline__ float fdec(unsigned v) {
  return __uint_as_float((v & 0x80000000u) ? (v ^ 0x80000000u) : ~v);
}

// ---- generic fill ---------------------------------------------------------
__global__ void fill_u32(unsigned* p, unsigned v, long n) {
  long i = (long)blockIdx.x * blockDim.x + threadIdx.x;
  if (i < n) p[i] = v;
}

// ---- layer1 linear: K=4, compute xl and xr together -----------------------
__global__ void lin1_kernel(const float* __restrict__ x,
                            const float* __restrict__ Wl, const float* __restrict__ bl,
                            const float* __restrict__ Wr, const float* __restrict__ br,
                            float* __restrict__ XL, float* __restrict__ XR, int N) {
  long idx = (long)blockIdx.x * blockDim.x + threadIdx.x;
  if (idx >= (long)N * HD) return;
  int n = (int)(idx >> 8), c = (int)(idx & 255);
  float4 xv = *(const float4*)(x + (size_t)n * 4);
  float sl = bl[c] + xv.x * Wl[c] + xv.y * Wl[HD + c] + xv.z * Wl[2 * HD + c] + xv.w * Wl[3 * HD + c];
  float sr = br[c] + xv.x * Wr[c] + xv.y * Wr[HD + c] + xv.z * Wr[2 * HD + c] + xv.w * Wr[3 * HD + c];
  XL[idx] = sl;
  XR[idx] = sr;
}

// ---- weight transpose + f32->f16: W[K][Nw] -> WT[Nw][K] -------------------
__global__ void convT_kernel(const float* __restrict__ W, fp16* __restrict__ WT, int K, int Nw) {
  long idx = (long)blockIdx.x * blockDim.x + threadIdx.x;
  if (idx >= (long)K * Nw) return;
  int n = (int)(idx / K), k = (int)(idx % K);
  WT[idx] = (fp16)W[(size_t)k * Nw + n];
}

__global__ void cvt16_kernel(const float* __restrict__ in, fp16* __restrict__ out, long n) {
  long i = (long)blockIdx.x * blockDim.x + threadIdx.x;
  if (i < n) out[i] = (fp16)in[i];
}

// ---- WMMA GEMM: C[M,Nmat] = A_f16[M,K] @ BT_f16[Nmat,K]^T + bias ----------
// one wave32 per 16x64 output tile: A fragment loaded once per K-step and
// reused across 4 column sub-tiles (4 accumulators, 2.5 b128 loads per WMMA)
__global__ void wmma_gemm64(const fp16* __restrict__ A, const fp16* __restrict__ BT,
                            const float* __restrict__ bias, float* __restrict__ C,
                            int M, int K, int Nmat, int do_relu) {
  int wave = blockIdx.x * (blockDim.x >> 5) + (threadIdx.x >> 5);
  int lane = threadIdx.x & 31;
  int tilesN = Nmat >> 6;                    // 64-wide column groups
  int tiles  = (M >> 4) * tilesN;
  if (wave >= tiles) return;
  int tm = wave / tilesN, tn = wave % tilesN;
  int l  = lane & 15;
  int hi = lane >> 4;                        // lane half selects K sub-range
  // A frag (16x32 f16): lanes 0-15 row=l, elems 0-15 K=k0..+15
  //                     lanes 16-31 row=l, elems 0-15 K=k0+16..+31
  const fp16* arow = A  + (size_t)(tm * 16 + l) * K + hi * 16;
  // B frag (32x16 f16): lane holds column n, K=k0..k0+15 (lanes 0-15) / +16..31 (lanes 16-31)
  const fp16* brow = BT + (size_t)(tn * 64 + l) * K + hi * 16;
  const size_t bstride = (size_t)16 * K;     // next 16-column sub-tile
  f8v acc0 = {}, acc1 = {}, acc2 = {}, acc3 = {};
  for (int k0 = 0; k0 < K; k0 += 32) {
    h8v a_lo = *(const h8v*)(arow + k0);
    h8v a_hi = *(const h8v*)(arow + k0 + 8);
    h16v a = __builtin_shufflevector(a_lo, a_hi, 0,1,2,3,4,5,6,7,8,9,10,11,12,13,14,15);
    h16v b0 = *(const h16v*)(brow + k0);
    h16v b1 = *(const h16v*)(brow + bstride + k0);
    h16v b2 = *(const h16v*)(brow + 2 * bstride + k0);
    h16v b3 = *(const h16v*)(brow + 3 * bstride + k0);
    acc0 = __builtin_amdgcn_wmma_f32_16x16x32_f16(false, a, false, b0, (short)0, acc0, false, false);
    acc1 = __builtin_amdgcn_wmma_f32_16x16x32_f16(false, a, false, b1, (short)0, acc1, false, false);
    acc2 = __builtin_amdgcn_wmma_f32_16x16x32_f16(false, a, false, b2, (short)0, acc2, false, false);
    acc3 = __builtin_amdgcn_wmma_f32_16x16x32_f16(false, a, false, b3, (short)0, acc3, false, false);
  }
  int row0 = tm * 16 + hi * 8;               // C/D: vgpr v -> row v (lanes 0-15) / v+8 (lanes 16-31)
  f8v accs[4] = {acc0, acc1, acc2, acc3};
  #pragma unroll
  for (int j = 0; j < 4; ++j) {
    int col  = tn * 64 + j * 16 + l;
    float bv = bias ? bias[col] : 0.0f;
    #pragma unroll
    for (int v = 0; v < 8; ++v) {
      float r = accs[j][v] + bv;
      if (do_relu) r = fmaxf(r, 0.0f);
      C[(size_t)(row0 + v) * Nmat + col] = r;
    }
  }
}

// ---- edge phase 1: attention logits + segment max -------------------------
__global__ void edge_s_kernel(const int* __restrict__ ei, const float* __restrict__ ea,
                              const float* __restrict__ XL, const float* __restrict__ XR,
                              const float* __restrict__ We, const float* __restrict__ att,
                              float* __restrict__ S, unsigned* __restrict__ SMAX, int E) {
  long t = (long)blockIdx.x * blockDim.x + threadIdx.x;
  if (t >= (long)E * NHEAD) return;
  int e = (int)(t >> 2), h = (int)(t & 3);
  int src = ei[e], dst = ei[E + e];
  float eav = ea[e];
  const float4* xl4 = (const float4*)(XL + (size_t)src * HD + h * 64);
  const float4* xr4 = (const float4*)(XR + (size_t)dst * HD + h * 64);
  const float4* we4 = (const float4*)(We + h * 64);
  const float4* at4 = (const float4*)(att + h * 64);
  float s = 0.f;
  #pragma unroll
  for (int i = 0; i < 16; ++i) {
    float4 a = xl4[i], b = xr4[i], w = we4[i], at = at4[i];
    float m;
    m = a.x + b.x + eav * w.x; m = (m > 0.f) ? m : NEG_SLOPE * m; s += m * at.x;
    m = a.y + b.y + eav * w.y; m = (m > 0.f) ? m : NEG_SLOPE * m; s += m * at.y;
    m = a.z + b.z + eav * w.z; m = (m > 0.f) ? m : NEG_SLOPE * m; s += m * at.z;
    m = a.w + b.w + eav * w.w; m = (m > 0.f) ? m : NEG_SLOPE * m; s += m * at.w;
  }
  S[t] = s;
  atomicMax(&SMAX[(size_t)dst * NHEAD + h], fenc(s));
}

// ---- edge phase 2: a = exp(s - smax[dst]); denom += a ---------------------
__global__ void edge_a_kernel(const int* __restrict__ ei, float* __restrict__ S,
                              const unsigned* __restrict__ SMAX, float* __restrict__ DEN, int E) {
  long t = (long)blockIdx.x * blockDim.x + threadIdx.x;
  if (t >= (long)E * NHEAD) return;
  int e = (int)(t >> 2), h = (int)(t & 3);
  int dst = ei[E + e];
  float smax = fdec(SMAX[(size_t)dst * NHEAD + h]);
  float a = expf(S[t] - smax);
  S[t] = a;
  atomicAdd(&DEN[(size_t)dst * NHEAD + h], a);
}

// ---- edge phase 3: ACC[dst] += (a/denom) * XL[src], float4 granularity ----
__global__ void edge_scatter_kernel(const int* __restrict__ ei, const float* __restrict__ S,
                                    const float* __restrict__ DEN, const float* __restrict__ XL,
                                    float* __restrict__ ACC, int E) {
  long t = (long)blockIdx.x * blockDim.x + threadIdx.x;
  if (t >= (long)E * 64) return;
  int e = (int)(t >> 6), c4 = (int)(t & 63);
  int src = ei[e], dst = ei[E + e];
  int h = c4 >> 4;
  float a   = S[(size_t)e * NHEAD + h];
  float den = DEN[(size_t)dst * NHEAD + h] + 1e-16f;
  float alpha = a / den;
  float4 xv = *(const float4*)(XL + (size_t)src * HD + c4 * 4);
  float* o = ACC + (size_t)dst * HD + c4 * 4;
  atomicAdd(o + 0, alpha * xv.x);
  atomicAdd(o + 1, alpha * xv.y);
  atomicAdd(o + 2, alpha * xv.z);
  atomicAdd(o + 3, alpha * xv.w);
}

// ---- H = relu(ACC + bias) -------------------------------------------------
__global__ void relu_bias_kernel(const float* __restrict__ ACC, const float* __restrict__ bias,
                                 float* __restrict__ H, long total) {
  long i = (long)blockIdx.x * blockDim.x + threadIdx.x;
  if (i >= total) return;
  H[i] = fmaxf(ACC[i] + bias[(int)(i & 255)], 0.0f);
}

// ---- global mean pool: scatter sums + counts ------------------------------
__global__ void pool_kernel(const float* __restrict__ H, const int* __restrict__ bidx,
                            float* __restrict__ SUMS, float* __restrict__ CNTS, int N) {
  long t = (long)blockIdx.x * blockDim.x + threadIdx.x;
  if (t >= (long)N * 64) return;
  int n = (int)(t >> 6), c4 = (int)(t & 63);
  int b = bidx[n];
  float4 v = *(const float4*)(H + (size_t)n * HD + c4 * 4);
  float* o = SUMS + (size_t)b * HD + c4 * 4;
  atomicAdd(o + 0, v.x);
  atomicAdd(o + 1, v.y);
  atomicAdd(o + 2, v.z);
  atomicAdd(o + 3, v.w);
  if (c4 == 0) atomicAdd(&CNTS[b], 1.0f);
}

__global__ void feats_kernel(const float* __restrict__ SUMS, const float* __restrict__ CNTS,
                             fp16* __restrict__ F16, int B) {
  long idx = (long)blockIdx.x * blockDim.x + threadIdx.x;
  if (idx >= (long)B * HD) return;
  int b = (int)(idx >> 8);
  float c = fmaxf(CNTS[b], 1.0f);
  F16[idx] = (fp16)(SUMS[idx] / c);
}

// ---- tiny dense layer (B=64 rows) -----------------------------------------
__global__ void dense_kernel(const float* __restrict__ in, const float* __restrict__ W,
                             const float* __restrict__ bv, float* __restrict__ out,
                             int M, int K, int Nn, int do_relu) {
  long idx = (long)blockIdx.x * blockDim.x + threadIdx.x;
  if (idx >= (long)M * Nn) return;
  int m = (int)(idx / Nn), n = (int)(idx % Nn);
  float s = bv[n];
  for (int k = 0; k < K; ++k) s += in[(size_t)m * K + k] * W[(size_t)k * Nn + n];
  if (do_relu) s = fmaxf(s, 0.0f);
  out[idx] = s;
}

// ---- log_softmax / gather / entropy: one block per batch row --------------
__global__ void head_kernel(const float* __restrict__ out_logits, const int* __restrict__ action,
                            float* __restrict__ out, int B, int A) {
  __shared__ float red[128];
  int b = blockIdx.x, t = threadIdx.x;
  float lg = (t < A) ? out_logits[(size_t)b * A + t] : -3.4e38f;
  red[t] = lg; __syncthreads();
  for (int s = 64; s > 0; s >>= 1) { if (t < s) red[t] = fmaxf(red[t], red[t + s]); __syncthreads(); }
  float mx = red[0]; __syncthreads();
  float ex = (t < A) ? expf(lg - mx) : 0.0f;
  red[t] = ex; __syncthreads();
  for (int s = 64; s > 0; s >>= 1) { if (t < s) red[t] += red[t + s]; __syncthreads(); }
  float lse = logf(red[0]) + mx; __syncthreads();
  float lp = lg - lse;
  red[t] = (t < A) ? (-expf(lp) * lp) : 0.0f; __syncthreads();
  for (int s = 64; s > 0; s >>= 1) { if (t < s) red[t] += red[t + s]; __syncthreads(); }
  if (t == 0) {
    out[(size_t)B * A + b]     = out_logits[(size_t)b * A + action[b]] - lse;  // log_prob
    out[(size_t)B * A + B + b] = red[0];                                       // entropy
  }
}

// ===========================================================================
extern "C" void kernel_launch(void* const* d_in, const int* in_sizes, int n_in,
                              void* d_out, int out_size, void* d_ws, size_t ws_size,
                              hipStream_t stream) {
  const float* x    = (const float*)d_in[0];
  const int*   ei   = (const int*)d_in[1];
  const float* ea   = (const float*)d_in[2];
  const int*   bidx = (const int*)d_in[3];
  const int*   act  = (const int*)d_in[4];
  const float *Wl1 = (const float*)d_in[5],  *bl1 = (const float*)d_in[6];
  const float *Wr1 = (const float*)d_in[7],  *br1 = (const float*)d_in[8];
  const float *We1 = (const float*)d_in[9],  *att1 = (const float*)d_in[10], *b1 = (const float*)d_in[11];
  const float *Wl2 = (const float*)d_in[12], *bl2 = (const float*)d_in[13];
  const float *Wr2 = (const float*)d_in[14], *br2 = (const float*)d_in[15];
  const float *We2 = (const float*)d_in[16], *att2 = (const float*)d_in[17], *b2 = (const float*)d_in[18];
  const float *Wc1 = (const float*)d_in[19], *bc1 = (const float*)d_in[20];
  const float *Wc2 = (const float*)d_in[21], *bc2 = (const float*)d_in[22];
  const float *Wc3 = (const float*)d_in[23], *bc3 = (const float*)d_in[24];
  const float *Wa1 = (const float*)d_in[25], *ba1 = (const float*)d_in[26];
  const float *Wa2 = (const float*)d_in[27], *ba2 = (const float*)d_in[28];
  const float *Wa3 = (const float*)d_in[29], *ba3 = (const float*)d_in[30];

  const int N = in_sizes[0] / 4;
  const int E = in_sizes[2];
  const int B = in_sizes[4];
  const int A = 100;

  // workspace carve-up (all 16B-aligned)
  float*    XL   = (float*)d_ws;                       // N*256  (xl)
  float*    XR   = XL + (size_t)N * HD;                // N*256  (xr, reused as ACC)
  float*    H    = XR + (size_t)N * HD;                // N*256
  fp16*     H16  = (fp16*)(H + (size_t)N * HD);        // N*256 f16
  float*    S    = (float*)(H16 + (size_t)N * HD);     // E*4
  unsigned* SMAX = (unsigned*)(S + (size_t)E * NHEAD); // N*4
  float*    DEN  = (float*)(SMAX + (size_t)N * NHEAD); // N*4
  fp16*     WL2T = (fp16*)(DEN + (size_t)N * NHEAD);   // 256*256
  fp16*     WR2T = WL2T + 65536;                       // 256*256
  fp16*     WC1T = WR2T + 65536;                       // 128*256
  fp16*     WA1T = WC1T + 32768;                       // 128*256
  float*    SUMS = (float*)(WA1T + 32768);             // B*256
  float*    CNTS = SUMS + (size_t)B * HD;              // B
  fp16*     F16  = (fp16*)(CNTS + B);                  // B*256 f16
  float*    HA1  = (float*)(F16 + (size_t)B * HD);     // B*128
  float*    HC1  = HA1 + (size_t)B * 128;              // B*128
  float*    HA2  = HC1 + (size_t)B * 128;              // B*64
  float*    HC2  = HA2 + (size_t)B * 64;               // B*64
  float*    out  = (float*)d_out;                      // [B*A | B | B | B]

  auto g = [](long total, int bs) { return dim3((unsigned)((total + bs - 1) / bs)); };
  const int BS = 256;

  // f16-transposed weights for WMMA consumers
  convT_kernel<<<g(65536, BS), BS, 0, stream>>>(Wl2, WL2T, 256, 256);
  convT_kernel<<<g(65536, BS), BS, 0, stream>>>(Wr2, WR2T, 256, 256);
  convT_kernel<<<g(32768, BS), BS, 0, stream>>>(Wc1, WC1T, 256, 128);
  convT_kernel<<<g(32768, BS), BS, 0, stream>>>(Wa1, WA1T, 256, 128);

  // ---------------- GATv2 layer 1 ----------------
  lin1_kernel<<<g((long)N * HD, BS), BS, 0, stream>>>(x, Wl1, bl1, Wr1, br1, XL, XR, N);
  fill_u32<<<g((long)N * NHEAD, BS), BS, 0, stream>>>(SMAX, 0u, (long)N * NHEAD);
  fill_u32<<<g((long)N * NHEAD, BS), BS, 0, stream>>>((unsigned*)DEN, 0u, (long)N * NHEAD);
  edge_s_kernel<<<g((long)E * NHEAD, BS), BS, 0, stream>>>(ei, ea, XL, XR, We1, att1, S, SMAX, E);
  edge_a_kernel<<<g((long)E * NHEAD, BS), BS, 0, stream>>>(ei, S, SMAX, DEN, E);
  fill_u32<<<g((long)N * HD, BS), BS, 0, stream>>>((unsigned*)XR, 0u, (long)N * HD);  // XR -> ACC
  edge_scatter_kernel<<<g((long)E * 64, BS), BS, 0, stream>>>(ei, S, DEN, XL, XR, E);
  relu_bias_kernel<<<g((long)N * HD, BS), BS, 0, stream>>>(XR, b1, H, (long)N * HD);  // h1

  // ---------------- layer-2 transforms via WMMA ----------------
  cvt16_kernel<<<g((long)N * HD, BS), BS, 0, stream>>>(H, H16, (long)N * HD);
  {
    long tiles = ((long)N / 16) * (256 / 64);        // 16x64 per wave
    wmma_gemm64<<<g(tiles * 32, BS), BS, 0, stream>>>(H16, WL2T, bl2, XL, N, 256, 256, 0);
    wmma_gemm64<<<g(tiles * 32, BS), BS, 0, stream>>>(H16, WR2T, br2, XR, N, 256, 256, 0);
  }

  // ---------------- GATv2 layer 2 ----------------
  fill_u32<<<g((long)N * NHEAD, BS), BS, 0, stream>>>(SMAX, 0u, (long)N * NHEAD);
  fill_u32<<<g((long)N * NHEAD, BS), BS, 0, stream>>>((unsigned*)DEN, 0u, (long)N * NHEAD);
  edge_s_kernel<<<g((long)E * NHEAD, BS), BS, 0, stream>>>(ei, ea, XL, XR, We2, att2, S, SMAX, E);
  edge_a_kernel<<<g((long)E * NHEAD, BS), BS, 0, stream>>>(ei, S, SMAX, DEN, E);
  fill_u32<<<g((long)N * HD, BS), BS, 0, stream>>>((unsigned*)XR, 0u, (long)N * HD);
  edge_scatter_kernel<<<g((long)E * 64, BS), BS, 0, stream>>>(ei, S, DEN, XL, XR, E);
  relu_bias_kernel<<<g((long)N * HD, BS), BS, 0, stream>>>(XR, b2, H, (long)N * HD);  // h2

  // ---------------- mean pool ----------------
  fill_u32<<<g((long)B * HD, BS), BS, 0, stream>>>((unsigned*)SUMS, 0u, (long)B * HD);
  fill_u32<<<g(B, 64), 64, 0, stream>>>((unsigned*)CNTS, 0u, B);
  pool_kernel<<<g((long)N * 64, BS), BS, 0, stream>>>(H, bidx, SUMS, CNTS, N);
  feats_kernel<<<g((long)B * HD, BS), BS, 0, stream>>>(SUMS, CNTS, F16, B);

  // ---------------- heads: first layers via WMMA, rest tiny ----------------
  {
    long tiles = ((long)B / 16) * (128 / 64);        // 16x64 per wave
    wmma_gemm64<<<g(tiles * 32, BS), BS, 0, stream>>>(F16, WA1T, ba1, HA1, B, 256, 128, 1);
    wmma_gemm64<<<g(tiles * 32, BS), BS, 0, stream>>>(F16, WC1T, bc1, HC1, B, 256, 128, 1);
  }
  dense_kernel<<<g((long)B * 64, BS), BS, 0, stream>>>(HA1, Wa2, ba2, HA2, B, 128, 64, 1);
  dense_kernel<<<g((long)B * 64, BS), BS, 0, stream>>>(HC1, Wc2, bc2, HC2, B, 128, 64, 1);
  dense_kernel<<<g((long)B * A, BS), BS, 0, stream>>>(HA2, Wa3, ba3, out, B, 64, A, 0);          // logits -> d_out
  dense_kernel<<<g(B, 64), 64, 0, stream>>>(HC2, Wc3, bc3, out + (size_t)B * A + 2 * B, B, 64, 1, 0); // value
  head_kernel<<<dim3(B), dim3(128), 0, stream>>>(out, act, out, B, A);
}
